// DiffeqSolver_42666205118907
// MI455X (gfx1250) — compile-verified, matrix operand order checked by reading, not run on hardware
//
#include <hip/hip_runtime.h>
#include <hip/hip_bf16.h>

// CDNA5 (gfx1250) fused graph-ODE kernel.
//  - one workgroup per (batch, receiver-node): edges r*49..r*49+48 are contiguous
//  - edge MLP (2x [64x128]x[128x128] GEMMs, both edge types) on v_wmma_f32_16x16x32_f16
//  - per-type weight blob (32 KB) staged into LDS with the Tensor Data Mover
//    (tensor_load_to_lds inline asm + s_wait_tensorcnt), overlaid on the dead
//    pre-gather LDS region; B fragments then come from LDS (ds_load_b128)
//  - type gating + edge->node mean reduced in LDS (deterministic, no global atomics)
//  - fused agg@W3 + b3 + tanh GEMV at block end
//  - RK4 orchestrated host-side; dt read from device time_steps

typedef __attribute__((ext_vector_type(16))) _Float16 v16h;
typedef __attribute__((ext_vector_type(8)))  float    v8f;
typedef unsigned int u32;
typedef __attribute__((ext_vector_type(4))) u32 u32x4;
typedef __attribute__((ext_vector_type(8))) u32 u32x8;

namespace {
constexpr int NATOMS = 50;
constexpr int FDIM   = 64;
constexpr int HDIM   = 128;
constexpr int KTYPES = 2;
constexpr int DEG    = NATOMS - 1;            // 49 edges per receiver
constexpr int EDGES  = NATOMS * DEG;          // 2450
constexpr int BATCH  = 64;                    // S*b
constexpr int TSTEPS = 40;
constexpr int TE     = 64;                    // edge slots per block (pad 49 -> 64)
constexpr int YTOT   = BATCH * NATOMS * FDIM; // 204800 floats
constexpr int WKHALF = 4 * 8 * 512;           // halves per (weight, edge-type) blob = 16384
constexpr int WFRAG  = KTYPES * WKHALF;       // 32768 halves per packed weight
}

// K-index mapping for 16-bit WMMA A/B fragments (ISA 7.12.2):
// lanes 0-15 hold K {0..7,16..23}, lanes 16-31 hold K {8..15,24..31}
__device__ __forceinline__ int kmap(int lane, int i) {
  int kb = (lane >> 4) * 8;
  return (i < 8) ? (kb + i) : (16 + kb + (i - 8));
}

// TDM 1-D copy: ndwords dwords from global gsrc into LDS at byte offset lds_off.
// D# group0/group1 packed per CDNA5 ISA 8.3/8.4 (2D form, VADDR2/3 omitted).
__device__ __forceinline__ void tdm_load_1d(const void* gsrc, u32 lds_off, u32 ndwords) {
  unsigned long long ga = (unsigned long long)(size_t)gsrc;
  u32x4 g0;
  g0[0] = 1u;                                            // count=1 valid user D#
  g0[1] = lds_off;                                       // lds_addr (bytes)
  g0[2] = (u32)ga;                                       // global_addr[31:0]
  g0[3] = ((u32)(ga >> 32) & 0x01FFFFFFu) | (2u << 30);  // global_addr[56:32] | type=2
  u32x8 g1;
  g1[0] = (2u << 16);                                    // data_size = 4 bytes
  g1[1] = (ndwords & 0xFFFFu) << 16;                     // tensor_dim0[15:0]  @ bits 63:48
  g1[2] = ((ndwords >> 16) & 0xFFFFu) | (1u << 16);      // tensor_dim0[31:16] | tensor_dim1=1
  g1[3] = (ndwords & 0xFFFFu) << 16;                     // tile_dim0 @ bits 127:112
  g1[4] = 0u;                                            // tile_dim1/2 unused
  g1[5] = ndwords;                                       // tensor_dim0_stride[31:0]
  g1[6] = 0u;
  g1[7] = 0u;
  asm volatile("tensor_load_to_lds %0, %1" :: "s"(g0), "s"(g1) : "memory");
}

// Pack W [K][128][128] fp32 -> fragment-ordered fp16 so a lane's B-fragment is
// one contiguous 32B chunk: out[((k*4+kt)*8+nt)*512 + lane*16 + i]
__global__ __launch_bounds__(256) void pack_weights_kernel(const float* __restrict__ W,
                                                           _Float16* __restrict__ out) {
  int idx = blockIdx.x * blockDim.x + threadIdx.x;
  if (idx >= WFRAG) return;
  int i    = idx & 15;
  int lane = (idx >> 4) & 31;
  int nt   = (idx >> 9) & 7;
  int kt   = (idx >> 12) & 3;
  int k    = (idx >> 14) & 1;
  int K = kt * 32 + kmap(lane, i);
  int n = nt * 16 + (lane & 15);
  out[idx] = (_Float16)W[((size_t)k * 128 + K) * 128 + n];
}

// One block = (batch b0, receiver r). 128 threads = 4 waves, wave w owns edge
// rows [w*16, w*16+16).
__global__ __launch_bounds__(128)
void ode_kernel(const float* __restrict__ y, const int* __restrict__ graph,
                const _Float16* __restrict__ w1f, const _Float16* __restrict__ w2f,
                const float* __restrict__ b1, const float* __restrict__ b2,
                const float* __restrict__ W3, const float* __restrict__ b3,
                float* __restrict__ dy) {
  // 32 KB region: first used as preLDS (16 KB gather of [send||recv] fp16),
  // then (after afrag capture) reused as the TDM weight staging buffer.
  __shared__ __align__(128) char smemA[WKHALF * 2];      // 32 KB
  __shared__ _Float16 hLDS[4][16][HDIM];                 // 16 KB per-wave hidden tile
  __shared__ float    aggP[8][HDIM];                     // 4 KB partial edge sums
  __shared__ int      typeLDS[TE];

  _Float16 (*preLDS)[2 * FDIM] = reinterpret_cast<_Float16 (*)[2 * FDIM]>(smemA);
  _Float16* wbuf               = reinterpret_cast<_Float16*>(smemA);
  const u32 wbufOff            = (u32)(size_t)(void*)smemA;   // LDS byte offset

  const int tid = threadIdx.x;
  const int b0  = blockIdx.x / NATOMS;
  const int r   = blockIdx.x % NATOMS;
  const float* yb = y + (size_t)b0 * NATOMS * FDIM;

  // ---- phase 0: gather pre = [send || recv] (f32 -> f16) and edge types ----
  if (tid < TE)
    typeLDS[tid] = (tid < DEG) ? graph[(size_t)b0 * EDGES + r * DEG + tid] : -1;

  const float4* yb4 = reinterpret_cast<const float4*>(yb);
  for (int q = tid; q < TE * 32; q += 128) {   // 32 float4 per 128-wide row
    int j  = q >> 5;
    int c4 = q & 31;
    float4 v = make_float4(0.0f, 0.0f, 0.0f, 0.0f);
    if (j < DEG) {
      int node = (c4 < 16) ? (j + (j >= r ? 1 : 0)) : r;   // sender skips diagonal
      v = yb4[node * 16 + (c4 & 15)];
    }
    _Float16* dst = &preLDS[j][c4 * 4];
    dst[0] = (_Float16)v.x; dst[1] = (_Float16)v.y;
    dst[2] = (_Float16)v.z; dst[3] = (_Float16)v.w;
  }
  __syncthreads();

  const int lane = tid & 31;
  const int w    = tid >> 5;
  const int lh   = lane >> 4;   // which half of the wave
  const int ln   = lane & 15;

  // edge types for this lane's 8 accumulator rows (C layout: M = rr + 8*lh)
  int etype[8];
#pragma unroll
  for (int rr = 0; rr < 8; ++rr) etype[rr] = typeLDS[w * 16 + lh * 8 + rr];

  const v8f zero8 = {};

  // A-fragments of pre (shared by both edge types): 4 K-tiles of 16x32 fp16
  v16h afrag[4];
#pragma unroll
  for (int kt = 0; kt < 4; ++kt) {
#pragma unroll
    for (int i = 0; i < 16; ++i)
      afrag[kt][i] = preLDS[w * 16 + ln][kt * 32 + kmap(lane, i)];
  }
  __syncthreads();   // preLDS dead; smemA may be reused as weight buffer

  v8f msel[8];
#pragma unroll
  for (int nt = 0; nt < 8; ++nt) msel[nt] = zero8;

  for (int k = 0; k < KTYPES; ++k) {
    // ---- stage W1[k] (32 KB) into LDS via Tensor Data Mover ----
    if (tid < 32) {
      tdm_load_1d(w1f + (size_t)k * WKHALF, wbufOff, WKHALF / 2);
      __builtin_amdgcn_s_wait_tensorcnt(0);
    }
    __syncthreads();

    // ---- GEMM1: h = relu(pre @ W1[k] + b1[k]) ----
    v8f acc[8];
#pragma unroll
    for (int nt = 0; nt < 8; ++nt) acc[nt] = zero8;
#pragma unroll
    for (int nt = 0; nt < 8; ++nt) {
#pragma unroll
      for (int kt = 0; kt < 4; ++kt) {
        v16h bf = *(const v16h*)(wbuf + (kt * 8 + nt) * 512 + lane * 16);
        acc[nt] = __builtin_amdgcn_wmma_f32_16x16x32_f16(
            false, afrag[kt], false, bf, (short)0, acc[nt], false, false);
      }
    }
#pragma unroll
    for (int nt = 0; nt < 8; ++nt) {
      float bias = b1[k * HDIM + nt * 16 + ln];
#pragma unroll
      for (int rr = 0; rr < 8; ++rr) {
        float hv = acc[nt][rr] + bias;
        hv = hv > 0.0f ? hv : 0.0f;
        hLDS[w][rr + lh * 8][nt * 16 + ln] = (_Float16)hv;
      }
    }
    __syncthreads();   // all wbuf reads done, hLDS published

    // ---- stage W2[k] into the same LDS buffer; re-fragment h meanwhile ----
    if (tid < 32) {
      tdm_load_1d(w2f + (size_t)k * WKHALF, wbufOff, WKHALF / 2);
    }
    v16h ah[4];
#pragma unroll
    for (int kt = 0; kt < 4; ++kt) {
#pragma unroll
      for (int i = 0; i < 16; ++i)
        ah[kt][i] = hLDS[w][ln][kt * 32 + kmap(lane, i)];
    }
    if (tid < 32) {
      __builtin_amdgcn_s_wait_tensorcnt(0);
    }
    __syncthreads();

    // ---- GEMM2: m = relu(h @ W2[k] + b2[k]), gated accumulate ----
    v8f acc2[8];
#pragma unroll
    for (int nt = 0; nt < 8; ++nt) acc2[nt] = zero8;
#pragma unroll
    for (int nt = 0; nt < 8; ++nt) {
#pragma unroll
      for (int kt = 0; kt < 4; ++kt) {
        v16h bf = *(const v16h*)(wbuf + (kt * 8 + nt) * 512 + lane * 16);
        acc2[nt] = __builtin_amdgcn_wmma_f32_16x16x32_f16(
            false, ah[kt], false, bf, (short)0, acc2[nt], false, false);
      }
    }
#pragma unroll
    for (int nt = 0; nt < 8; ++nt) {
      float bias = b2[k * HDIM + nt * 16 + ln];
#pragma unroll
      for (int rr = 0; rr < 8; ++rr) {
        float mv = acc2[nt][rr] + bias;
        mv = mv > 0.0f ? mv : 0.0f;
        if (etype[rr] == k) msel[nt][rr] += mv;
      }
    }
    __syncthreads();   // wbuf + hLDS reused next k
  }

  // ---- edge -> node reduction (deterministic LDS tree) ----
#pragma unroll
  for (int nt = 0; nt < 8; ++nt) {
    float s = 0.0f;
#pragma unroll
    for (int rr = 0; rr < 8; ++rr) s += msel[nt][rr];
    aggP[w * 2 + lh][nt * 16 + ln] = s;   // unique (partial, column) writer
  }
  __syncthreads();
  if (tid < HDIM) {
    float s = 0.0f;
#pragma unroll
    for (int p = 0; p < 8; ++p) s += aggP[p][tid];
    aggP[0][tid] = s;
  }
  __syncthreads();

  // ---- dy = tanh(agg/N @ W3 + b3) ----
  if (tid < FDIM) {
    const float inv_n = 1.0f / (float)NATOMS;
    float acc = 0.0f;
#pragma unroll 4
    for (int hh = 0; hh < HDIM; ++hh) acc += aggP[0][hh] * W3[hh * FDIM + tid];
    acc = acc * inv_n + b3[tid];
    dy[((size_t)b0 * NATOMS + r) * FDIM + tid] = tanhf(acc);
  }
}

// yCur = first_point ; out[:, t=0, :] = first_point
__global__ __launch_bounds__(256)
void init_kernel(const float* __restrict__ fp, float* __restrict__ yCur,
                 float* __restrict__ out) {
  int idx = blockIdx.x * blockDim.x + threadIdx.x;
  if (idx >= YTOT) return;
  float v = fp[idx];
  yCur[idx] = v;
  int bn = idx >> 6, f = idx & 63;
  out[(size_t)bn * TSTEPS * FDIM + f] = v;
}

// yTmp = y + c*dt*k   (dt = ts[t]-ts[t-1], read from device)
__global__ __launch_bounds__(256)
void axpy_kernel(const float* __restrict__ y, const float* __restrict__ kk,
                 const float* __restrict__ ts, int t, float c,
                 float* __restrict__ out) {
  int idx = blockIdx.x * blockDim.x + threadIdx.x;
  if (idx >= YTOT) return;
  float dt = ts[t] - ts[t - 1];
  out[idx] = y[idx] + c * dt * kk[idx];
}

// y += dt/6*(k1+2k2+2k3+k4); out[:, t, :] = y
__global__ __launch_bounds__(256)
void rk4_final_kernel(float* __restrict__ yCur, const float* __restrict__ k1,
                      const float* __restrict__ k2, const float* __restrict__ k3,
                      const float* __restrict__ k4, const float* __restrict__ ts,
                      int t, float* __restrict__ out) {
  int idx = blockIdx.x * blockDim.x + threadIdx.x;
  if (idx >= YTOT) return;
  float dt = ts[t] - ts[t - 1];
  float yn = yCur[idx] +
             (dt * (1.0f / 6.0f)) * (k1[idx] + 2.0f * k2[idx] + 2.0f * k3[idx] + k4[idx]);
  yCur[idx] = yn;
  int bn = idx >> 6, f = idx & 63;
  out[(size_t)bn * TSTEPS * FDIM + (size_t)t * FDIM + f] = yn;
}

extern "C" void kernel_launch(void* const* d_in, const int* in_sizes, int n_in,
                              void* d_out, int out_size, void* d_ws, size_t ws_size,
                              hipStream_t stream) {
  (void)in_sizes; (void)n_in; (void)out_size; (void)ws_size;
  const float* fp    = (const float*)d_in[0];
  const float* ts    = (const float*)d_in[1];
  const int*   graph = (const int*)d_in[2];
  const float* W1    = (const float*)d_in[3];
  const float* b1    = (const float*)d_in[4];
  const float* W2    = (const float*)d_in[5];
  const float* b2    = (const float*)d_in[6];
  const float* W3    = (const float*)d_in[7];
  const float* b3    = (const float*)d_in[8];
  float* out = (float*)d_out;

  // carve workspace (256B aligned)
  auto alignup = [](size_t x) { return (x + 255) & ~(size_t)255; };
  char* base = (char*)d_ws;
  size_t off = 0;
  float* yCur = (float*)(base + off); off = alignup(off + (size_t)YTOT * 4);
  float* yTmp = (float*)(base + off); off = alignup(off + (size_t)YTOT * 4);
  float* k1   = (float*)(base + off); off = alignup(off + (size_t)YTOT * 4);
  float* k2   = (float*)(base + off); off = alignup(off + (size_t)YTOT * 4);
  float* k3   = (float*)(base + off); off = alignup(off + (size_t)YTOT * 4);
  float* k4   = (float*)(base + off); off = alignup(off + (size_t)YTOT * 4);
  _Float16* w1f = (_Float16*)(base + off); off = alignup(off + (size_t)WFRAG * 2);
  _Float16* w2f = (_Float16*)(base + off); off = alignup(off + (size_t)WFRAG * 2);

  const int ethreads = 256;
  const int eblocks  = (YTOT + ethreads - 1) / ethreads;   // 800
  const dim3 odeGrid(BATCH * NATOMS);                      // 3200 blocks

  pack_weights_kernel<<<WFRAG / 256, 256, 0, stream>>>(W1, w1f);
  pack_weights_kernel<<<WFRAG / 256, 256, 0, stream>>>(W2, w2f);
  init_kernel<<<eblocks, ethreads, 0, stream>>>(fp, yCur, out);

  for (int t = 1; t < TSTEPS; ++t) {
    ode_kernel<<<odeGrid, 128, 0, stream>>>(yCur, graph, w1f, w2f, b1, b2, W3, b3, k1);
    axpy_kernel<<<eblocks, ethreads, 0, stream>>>(yCur, k1, ts, t, 0.5f, yTmp);
    ode_kernel<<<odeGrid, 128, 0, stream>>>(yTmp, graph, w1f, w2f, b1, b2, W3, b3, k2);
    axpy_kernel<<<eblocks, ethreads, 0, stream>>>(yCur, k2, ts, t, 0.5f, yTmp);
    ode_kernel<<<odeGrid, 128, 0, stream>>>(yTmp, graph, w1f, w2f, b1, b2, W3, b3, k3);
    axpy_kernel<<<eblocks, ethreads, 0, stream>>>(yCur, k3, ts, t, 1.0f, yTmp);
    ode_kernel<<<odeGrid, 128, 0, stream>>>(yTmp, graph, w1f, w2f, b1, b2, W3, b3, k4);
    rk4_final_kernel<<<eblocks, ethreads, 0, stream>>>(yCur, k1, k2, k3, k4, ts, t, out);
  }
}
